// LI2Model_31327491457690
// MI455X (gfx1250) — matrix operand order repaired and checked by reading, not compiled, not run on hardware
//
#include <hip/hip_runtime.h>
#include <stdint.h>

typedef __attribute__((ext_vector_type(16))) _Float16 v16h;
typedef __attribute__((ext_vector_type(8)))  float    v8f;
typedef uint32_t __attribute__((may_alias)) u32a;

// norse constants: DT*TAU_MEM_INV = 0.1, DT*TAU_SYN_INV = 0.2
#define K_MEM 0.1f
#define K_SYN 0.2f

union Frag { v16h h; u32a u[8]; _Float16 e[16]; };

__launch_bounds__(32)
__global__ void snn_wmma_kernel(const float* __restrict__ xin,
                                const float* __restrict__ conv_w,
                                const float* __restrict__ conv_b,
                                const float* __restrict__ fc_w,
                                const float* __restrict__ fc_b,
                                const int* __restrict__ nsteps_p,
                                float* __restrict__ out,
                                int B)
{
    // Only data that crosses lanes lives in LDS:
    __shared__ float G [1568];   // static conv drive   [16][98] (read-only in loop)
    __shared__ float Zs[1568];   // spikes              [16][98] (pool reads cross-lane)
    __shared__ __align__(4) _Float16 ZH[1536]; // f16 WMMA-A staging [16][96], cols>=72 stay 0

    const int lane  = threadIdx.x;      // 0..31
    const int n16   = lane & 15;
    const int bbase = blockIdx.x * 16;
    const int T     = nsteps_p[0];

    // Lane-private recurrent state in registers (idx = e*32+lane is lane-invariant
    // across steps, so no cross-lane access is ever needed for v/i state).
    float rv[49], ri[49];     // LIF  v,i  -> element idx  = e*32+lane (16x98)
    float rv1[36], ri1[36];   // LI1  v,i  -> element idx2 = e*32+lane (16x72)
    float v2[8],  i2[8];      // LI2 readout in WMMA C/D layout (lane=N, vgpr=M)

    #pragma unroll
    for (int e = 0; e < 49; ++e) { rv[e] = 0.f; ri[e] = 0.f; }
    #pragma unroll
    for (int e = 0; e < 36; ++e) { rv1[e] = 0.f; ri1[e] = 0.f; }
    #pragma unroll
    for (int r = 0; r < 8; ++r) { v2[r] = 0.f; i2[r] = 0.f; }

    for (int idx = lane; idx < 1536; idx += 32) ZH[idx] = (_Float16)0.f;

    // Prefetch this tile's input rows (gfx1250 global_prefetch path)
    {
        int bg = bbase + n16;
        if (bg < B) __builtin_prefetch(&xin[(size_t)bg * 100], 0, 1);
    }

    // ---- static drive: g = conv2d(x, w, VALID) + b (hoisted out of time loop)
    for (int e = 0; e < 49; ++e) {
        int idx = e * 32 + lane;           // 0..1567
        int b = idx / 98;  int p = idx - b * 98;
        int c = p / 49;    int q = p - c * 49;
        int y = q / 7;     int x = q - y * 7;
        int bg = bbase + b;
        float acc = conv_b[c];
        if (bg < B) {
            const float* xb = xin + (size_t)bg * 100;
            #pragma unroll
            for (int ky = 0; ky < 4; ++ky)
                #pragma unroll
                for (int kx = 0; kx < 4; ++kx)
                    acc += xb[(y + ky) * 10 + (x + kx)] * conv_w[c * 16 + ky * 4 + kx];
        }
        G[idx] = acc;
    }

    // ---- fc_w into WMMA B-frags (f16), once. B layout (K x N = 32x16):
    // lanes 0-15: N=lane, K = {2v|2v+8}..+1 per VGPR v; lanes 16-31: same N, K += 8.
    const int khalf = (lane < 16) ? 0 : 8;
    v16h barr[3];
    {
        #pragma unroll
        for (int cch = 0; cch < 3; ++cch) {
            Frag bf;
            int kb = cch * 32 + khalf;
            #pragma unroll
            for (int v = 0; v < 8; ++v) {
                int k0 = kb + ((v < 4) ? 2 * v : 2 * v + 8), k1 = k0 + 1;
                float w0 = (n16 < 10 && k0 < 72) ? fc_w[n16 * 72 + k0] : 0.f;
                float w1 = (n16 < 10 && k1 < 72) ? fc_w[n16 * 72 + k1] : 0.f;
                bf.e[2 * v]     = (_Float16)w0;
                bf.e[2 * v + 1] = (_Float16)w1;
            }
            barr[cch] = bf.h;
        }
    }
    const float bias = (n16 < 10) ? fc_b[n16] : 0.f;

    __syncthreads();

    const int hsel = (lane < 16) ? 0 : 8;  // A layout: half-wave K split
    const int m    = n16;                  // A row handled by this lane

    #pragma unroll 1
    for (int t = 0; t < T; ++t) {
        // ---- LIF: decay, spike on old state, inject static drive (regs + Zs store)
        #pragma unroll
        for (int e = 0; e < 49; ++e) {
            int idx = e * 32 + lane;
            float v = rv[e], i = ri[e];
            float vd = (1.0f - K_MEM) * v + K_MEM * i;
            bool  sp = vd > 1.0f;
            rv[e] = sp ? 0.0f : vd;              // reset to V_RESET=0 on spike
            ri[e] = (1.0f - K_SYN) * i + G[idx];
            Zs[idx] = sp ? 1.0f : 0.0f;
        }
        __syncthreads();

        // ---- MaxPool(2x2, stride 1) + LI1; stage LI voltage as f16 for WMMA A
        #pragma unroll
        for (int e = 0; e < 36; ++e) {
            int idx = e * 32 + lane;       // 0..1151
            int b = idx / 72;  int o = idx - b * 72;
            int c = o / 36;    int rr = o - c * 36;
            int y = rr / 6;    int x = rr - y * 6;
            int zb = b * 98 + c * 49 + y * 7 + x;
            float z4 = fmaxf(fmaxf(Zs[zb], Zs[zb + 1]), fmaxf(Zs[zb + 7], Zs[zb + 8]));
            float v = rv1[e], i = ri1[e];
            float vn = (1.0f - K_MEM) * v + K_MEM * i;
            rv1[e] = vn;
            ri1[e] = (1.0f - K_SYN) * i + z4;
            ZH[b * 96 + o] = (_Float16)vn;
        }
        __syncthreads();

        // ---- FC 16x72 @ 72x10 (+bias) via 3 chained v_wmma_f32_16x16x32_f16.
        // A layout (M x K = 16x32): lane m<16 holds row m, K = {2v|2v+8};
        // lane m+16 holds row m, K += 8. Pairs contiguous -> b32 LDS loads.
        v8f cacc;
        #pragma unroll
        for (int r = 0; r < 8; ++r) cacc[r] = bias;

        #pragma unroll
        for (int cch = 0; cch < 3; ++cch) {
            Frag a;
            #pragma unroll
            for (int v = 0; v < 8; ++v) {
                int koff = ((v < 4) ? 2 * v : 2 * v + 8) + hsel;
                a.u[v] = *(const u32a*)&ZH[m * 96 + cch * 32 + koff];
            }
            cacc = __builtin_amdgcn_wmma_f32_16x16x32_f16(false, a.h, false, barr[cch],
                                                          (short)0, cacc, false, false);
        }

        // ---- LI2 readout recurrence directly in the WMMA D layout
        #pragma unroll
        for (int r = 0; r < 8; ++r) {
            float vn = (1.0f - K_MEM) * v2[r] + K_MEM * i2[r];
            i2[r] = (1.0f - K_SYN) * i2[r] + cacc[r];
            v2[r] = vn;
        }
        __syncthreads();
    }

    // ---- store final readout voltage -> out[B,10] from D layout
    int mrow = (lane < 16) ? 0 : 8;
    if (bbase + 16 <= B) {
        // full tile: no per-store bounds checks -> no saveexec chains
        if (n16 < 10) {
            #pragma unroll
            for (int r = 0; r < 8; ++r)
                out[(size_t)(bbase + mrow + r) * 10 + n16] = v2[r];
        }
    } else if (n16 < 10) {
        #pragma unroll
        for (int r = 0; r < 8; ++r) {
            int bg = bbase + mrow + r;
            if (bg < B) out[(size_t)bg * 10 + n16] = v2[r];
        }
    }
}

extern "C" void kernel_launch(void* const* d_in, const int* in_sizes, int n_in,
                              void* d_out, int out_size, void* d_ws, size_t ws_size,
                              hipStream_t stream) {
    (void)n_in; (void)d_ws; (void)ws_size; (void)out_size;
    const float* x      = (const float*)d_in[0];
    const float* conv_w = (const float*)d_in[1];
    const float* conv_b = (const float*)d_in[2];
    const float* fc_w   = (const float*)d_in[3];
    const float* fc_b   = (const float*)d_in[4];
    const int*   nsteps = (const int*)d_in[5];
    float* out = (float*)d_out;

    int B = in_sizes[0] / 100;           // x is [B,1,10,10]
    int nblocks = (B + 15) / 16;         // one wave32 per 16-batch tile
    snn_wmma_kernel<<<nblocks, 32, 0, stream>>>(x, conv_w, conv_b, fc_w, fc_b,
                                                nsteps, out, B);
}
